// LagrangeNet_20753281975115
// MI455X (gfx1250) — matrix operand order repaired and checked by reading, not compiled, not run on hardware
//
#include <hip/hip_runtime.h>

// ---- problem constants (from reference) ----
#define BB    256
#define FF    1024
#define DOF   64
#define NCLS  1000
#define KK    (FF * DOF)       // 65536 reduction length
#define TT    63               // dof - 1 simplices
#define KSPLIT 16
#define KCHUNK (KK / KSPLIT)   // 4096 per block
#define NIT   (KCHUNK / 64)    // 64 iterations per block
#define LDS_STRIDE 72          // 64 halves + pad (bank-conflict-free, keeps 16B align)

typedef __attribute__((ext_vector_type(16))) _Float16 v16h;
typedef __attribute__((ext_vector_type(8)))  _Float16 v8h;
typedef __attribute__((ext_vector_type(4)))  _Float16 v4h;
typedef __attribute__((ext_vector_type(8)))  float    v8f;

__global__ void zero_kernel(float* __restrict__ p, int n) {
    int i = blockIdx.x * blockDim.x + threadIdx.x;
    if (i < n) p[i] = 0.0f;
}

// Build one 16x32 f16 A-fragment for this lane from the hat-function embedding,
// entirely in packed f16 math with prescaled constants. ISA A layout:
//   lanes 0-15 : elem e (0..7) -> K = e,      elem e (8..15) -> K = e+8   (16..23)
//   lanes 16-31: elem e (0..7) -> K = e+8,    elem e (8..15) -> K = e+16  (24..31)
// With u = s - p0 (lane/substep base folded in), c_e = {0..7, 16..23}:
//   emb[e] = mn + scale * max(0, min(1 + (u-c_e), 1 - (u-c_e)))
//          = max(mn, min(g + A1[e], A2[e] - g)),   g = scale*u
//   A1[e] = mn + scale*(1 - c_e),  A2[e] = mn + scale*(1 + c_e)
// 4 packed ops per element pair: pk_add, pk_add(neg), pk_min, pk_max.
__device__ __forceinline__ v16h build_a(float g32, float gLo, float gHi,
                                        const v16h& a1v, const v16h& a2v,
                                        const v16h& mnv) {
    // clamp keeps the f16 conversion fine-grained and maps the out-of-domain
    // sentinel to "emb == mn everywhere"
    const _Float16 g = (_Float16)fminf(fmaxf(g32, gLo), gHi);
    v16h t1 = g + a1v;                                   // v_pk_add_f16
    v16h t2 = a2v - g;                                   // v_pk_add_f16 (neg)
    v16h t3 = __builtin_elementwise_min(t1, t2);         // v_pk_min_num_f16
    return __builtin_elementwise_max(t3, mnv);           // v_pk_max_num_f16
}

__global__ __launch_bounds__(256)
void lagrange_wmma_kernel(const float* __restrict__ x,
                          const float* __restrict__ minx,
                          const float* __restrict__ maxx,
                          const float* __restrict__ W,
                          float* __restrict__ out) {
    // double-buffered B tile: 2 x 64 classes x 64 K halves (2 x 9.2 KB)
    __shared__ __align__(16) _Float16 Bs[2][64][LDS_STRIDE];

    const int tid   = threadIdx.x;
    const int wave  = tid >> 5;
    const int lane  = tid & 31;
    const int lan16 = lane & 15;
    const int half  = lane >> 4;          // 0: lanes 0-15, 1: lanes 16-31

    const int n_base  = blockIdx.x * 64;          // 64 classes per block (16 blocks, pad to 1024)
    const int k_begin = blockIdx.y * KCHUNK;      // K-split

    const float mn    = minx[0];
    const float mx    = maxx[0];
    const float scale = mx - mn;
    const float inv   = (float)TT / scale;        // s = (x - mn) * inv
    const float gLo   = -2.0f * scale;
    const float gHi   = 26.0f * scale;

    // prescaled constant vectors for the packed-f16 hat evaluation
    v16h a1v, a2v, mnv;
#pragma unroll
    for (int e = 0; e < 16; ++e) {
        const float c = (float)(e < 8 ? e : e + 8);
        a1v[e] = (_Float16)(mn + scale * (1.0f - c));
        a2v[e] = (_Float16)(mn + scale * (1.0f + c));
        mnv[e] = (_Float16)mn;
    }
    // per-substep bases: g = scale*s - sp0[ks]
    float sp0[2];
    sp0[0] = scale * (float)(half * 8);
    sp0[1] = scale * (float)(32 + half * 8);

    // two M fragments per wave -> block covers all M = 256
    const int row0 = wave * 16 + lan16;
    const int row1 = 128 + row0;

    v8f acc[2][4];
    const v8f vzero = {};
#pragma unroll
    for (int mi = 0; mi < 2; ++mi)
#pragma unroll
        for (int nf = 0; nf < 4; ++nf)
            acc[mi][nf] = vzero;

    // cooperative stage of W tile (64 classes x 64 K) -> LDS buffer as f16
    auto stage = [&](int itx) {
        const int k0s = k_begin + itx * 64;
        _Float16 (*buf)[LDS_STRIDE] = Bs[itx & 1];
#pragma unroll
        for (int q = 0; q < 4; ++q) {
            const int e    = tid * 16 + q * 4;       // 4096 f32 elements / 256 thr
            const int n    = e >> 6;
            const int kk   = e & 63;
            const int nrow = min(n_base + n, NCLS - 1);   // clamp pad rows (stores guarded)
            const float4 w4 = *reinterpret_cast<const float4*>(
                W + (size_t)nrow * KK + (size_t)(k0s + kk));
            v4h h4 = { (_Float16)w4.x, (_Float16)w4.y, (_Float16)w4.z, (_Float16)w4.w };
            *reinterpret_cast<v4h*>(&buf[n][kk]) = h4;
        }
    };

    stage(0);

    for (int it = 0; it < NIT; ++it) {
        __syncthreads();                 // stage(it) visible; compute(it-1) done
        if (it + 1 < NIT) stage(it + 1); // prefetch next tile into other buffer

        const _Float16 (*buf)[LDS_STRIDE] = Bs[it & 1];

        // --- per-lane hat-embedding state for this 64-wide f block ---
        const int f = (k_begin >> 6) + it;     // one feature per 64-K block
        const float xv0 = x[(size_t)row0 * FF + f];
        const float xv1 = x[(size_t)row1 * FF + f];
        const float s0 = (xv0 - mn) * inv;
        const float s1 = (xv1 - mn) * inv;
        // out-of-domain -> basis all zero -> emb == mn everywhere: push g far low
        const float g0 = (s0 >= 0.0f && s0 <= (float)TT) ? scale * s0 : -1.0e30f;
        const float g1 = (s1 >= 0.0f && s1 <= (float)TT) ? scale * s1 : -1.0e30f;

        // build all 4 A fragments (2 K-substeps x 2 M rows) up front so no
        // VALU writer of an A register sits in the WMMA hazard shadow
        v16h aF[2][2];
#pragma unroll
        for (int ks = 0; ks < 2; ++ks) {
            aF[ks][0] = build_a(g0 - sp0[ks], gLo, gHi, a1v, a2v, mnv);
            aF[ks][1] = build_a(g1 - sp0[ks], gLo, gHi, a1v, a2v, mnv);
        }

#pragma unroll
        for (int ks = 0; ks < 2; ++ks) {       // two K=32 sub-steps per 64-K tile
            // B layout (KxN striped): lane holds column n = lan16,
            // K = kb .. kb+7 then kb+16 .. kb+23, kb = ks*32 + half*8
            const int kb = ks * 32 + half * 8;
#pragma unroll
            for (int nf = 0; nf < 4; ++nf) {
                const int n = nf * 16 + lan16;
                const v8h lo = *reinterpret_cast<const v8h*>(&buf[n][kb]);
                const v8h hi = *reinterpret_cast<const v8h*>(&buf[n][kb + 16]);
                const v16h b = __builtin_shufflevector(lo, hi,
                    0, 1, 2, 3, 4, 5, 6, 7, 8, 9, 10, 11, 12, 13, 14, 15);
                acc[0][nf] = __builtin_amdgcn_wmma_f32_16x16x32_f16(
                    false, aF[ks][0], false, b, (short)0, acc[0][nf], false, false);
                acc[1][nf] = __builtin_amdgcn_wmma_f32_16x16x32_f16(
                    false, aF[ks][1], false, b, (short)0, acc[1][nf], false, false);
            }
        }
    }

    // --- accumulate partial K-chunk results: hardware f32 atomics ---
    // C/D layout: VGPR r -> M = r + half*8, N = lan16
#pragma unroll
    for (int mi = 0; mi < 2; ++mi) {
        const int mrow_base = (mi ? 128 : 0) + wave * 16 + half * 8;
#pragma unroll
        for (int nf = 0; nf < 4; ++nf) {
            const int ncol = n_base + nf * 16 + lan16;
            if (ncol < NCLS) {
#pragma unroll
                for (int r = 0; r < 8; ++r) {
                    unsafeAtomicAdd(&out[(size_t)(mrow_base + r) * NCLS + ncol],
                                    acc[mi][nf][r]);
                }
            }
        }
    }
}

extern "C" void kernel_launch(void* const* d_in, const int* in_sizes, int n_in,
                              void* d_out, int out_size, void* d_ws, size_t ws_size,
                              hipStream_t stream) {
    const float* x    = (const float*)d_in[0];
    const float* minx = (const float*)d_in[1];
    const float* maxx = (const float*)d_in[2];
    const float* W    = (const float*)d_in[3];
    float* out = (float*)d_out;

    const int n = out_size; // 256 * 1000
    hipLaunchKernelGGL(zero_kernel, dim3((n + 255) / 256), dim3(256), 0, stream, out, n);
    hipLaunchKernelGGL(lagrange_wmma_kernel, dim3(16, KSPLIT), dim3(256), 0, stream,
                       x, minx, maxx, W, out);
}
